// ComputeDistances_55740085568003
// MI455X (gfx1250) — compile-verified, matrix operand order checked by reading, not run on hardware
//
#include <hip/hip_runtime.h>
#include <math.h>

typedef __attribute__((ext_vector_type(16))) __bf16 v16bf;
typedef __attribute__((ext_vector_type(8)))  __bf16 v8bf;
typedef __attribute__((ext_vector_type(8)))  float  v8f;

// ---------- bf16 helpers (RNE via bit ops; no __bf16 arithmetic) ----------
__device__ __forceinline__ unsigned short bf16_bits(float x) {
    unsigned u = __builtin_bit_cast(unsigned, x);
    return (unsigned short)((u + 0x7FFFu + ((u >> 16) & 1u)) >> 16);
}
__device__ __forceinline__ float bf16_val(unsigned short h) {
    unsigned u = ((unsigned)h) << 16;
    return __builtin_bit_cast(float, u);
}
__device__ __forceinline__ __bf16 bits_to_bf16(unsigned short h) {
    return __builtin_bit_cast(__bf16, h);
}
__device__ __forceinline__ float bf_to_f(__bf16 b) {
    return bf16_val(__builtin_bit_cast(unsigned short, b));
}

// Assemble a 16x32 bf16 A/B fragment row per the CDNA5 ISA layout:
// lane (half,row): elements 0..7  = K = half*8 + {0..7}
//                  elements 8..15 = K = 16 + half*8 + {0..7}
// Both runs are 16B-aligned contiguous LDS reads (rows padded to 40 bf16 = 80B).
__device__ __forceinline__ v16bf make_frag(const __bf16* rowp, int k0) {
    v8bf r0 = *(const v8bf*)(rowp + k0);
    v8bf r1 = *(const v8bf*)(rowp + 16 + k0);
    v16bf f;
#pragma unroll
    for (int t = 0; t < 8; ++t) { f[t] = r0[t]; f[8 + t] = r1[t]; }
    return f;
}

// ---------- conversion passes ----------
// fp32 [R x Cc] -> transposed bf16 hi/lo [Cc x R]
__global__ __launch_bounds__(256)
void transpose_split_kernel(const float* __restrict__ in,
                            __bf16* __restrict__ hi, __bf16* __restrict__ lo,
                            int R, int Cc) {
    __shared__ float tile[32][33];
    const int cx = threadIdx.x & 31;
    const int ry = threadIdx.x >> 5;      // 0..7
    const int c0 = blockIdx.x * 32, r0 = blockIdx.y * 32;
#pragma unroll
    for (int i = 0; i < 32; i += 8)
        tile[ry + i][cx] = in[(size_t)(r0 + ry + i) * Cc + (c0 + cx)];
    __syncthreads();
#pragma unroll
    for (int i = 0; i < 32; i += 8) {
        float v = tile[cx][ry + i];
        size_t o = (size_t)(c0 + ry + i) * R + (r0 + cx);
        unsigned short h = bf16_bits(v);
        hi[o] = bits_to_bf16(h);
        lo[o] = bits_to_bf16(bf16_bits(v - bf16_val(h)));
    }
}

// fp32 -> bf16 hi/lo, same layout
__global__ __launch_bounds__(256)
void split_kernel(const float* __restrict__ in,
                  __bf16* __restrict__ hi, __bf16* __restrict__ lo, int total) {
    int i = blockIdx.x * blockDim.x + threadIdx.x;
    if (i >= total) return;
    float v = in[i];
    unsigned short h = bf16_bits(v);
    hi[i] = bits_to_bf16(h);
    lo[i] = bits_to_bf16(bf16_bits(v - bf16_val(h)));
}

// ---------- main WMMA GEMM ----------
// C(MxN) = A(MxK) * B(KxN).  A given as bf16 hi/lo [M][K] (k-contiguous, lda).
// B given PRE-TRANSPOSED as bf16 hi/lo [N][K] (k-contiguous, ldb).
// 256 threads = 8 waves (2x4). Block tile 64x256, wave tile 32x64, K-step 32.
// 24 WMMAs per wave per K-step (hi*hi + lo*hi + hi*lo, fp32 accumulate).
// mode 0: Cf[i,j] = acc                         (fp32)
// mode 1: Cf[i,j] = sqrt(max(qcol[j]-2acc+qrow[i],0))
// mode 2: Dhi/Dlo[i,j] = bf16 split of acc      (natural, ldc)
// mode 3: Dhi/Dlo[j,i] = bf16 split of acc      (transposed, ldc)
__global__ __launch_bounds__(256)
void gemm_bf16pair_wmma_kernel(const __bf16* __restrict__ Ahi, const __bf16* __restrict__ Alo, int lda,
                               const __bf16* __restrict__ Bhi, const __bf16* __restrict__ Blo, int ldb,
                               float* __restrict__ Cf,
                               __bf16* __restrict__ Dhi, __bf16* __restrict__ Dlo, int ldc,
                               int Kdim, int mode,
                               const float* __restrict__ qrow, const float* __restrict__ qcol) {
    const int tid  = threadIdx.x;
    const int wave = tid >> 5;           // 0..7
    const int wm   = wave >> 2;          // 0..1
    const int wn   = wave & 3;           // 0..3
    const int lane = tid & 31;
    const int half = lane >> 4;
    const int row  = lane & 15;

    const int tileM = blockIdx.y * 64;
    const int tileN = blockIdx.x * 256;

    // Rows padded to 40 bf16 (80B): 16B-aligned vector access, conflict-free banks.
    __shared__ __attribute__((aligned(16))) __bf16 lds_a_hi[64][40];
    __shared__ __attribute__((aligned(16))) __bf16 lds_a_lo[64][40];
    __shared__ __attribute__((aligned(16))) __bf16 lds_b_hi[256][40];
    __shared__ __attribute__((aligned(16))) __bf16 lds_b_lo[256][40];

    v8f acc[2][4];
#pragma unroll
    for (int mt = 0; mt < 2; ++mt)
#pragma unroll
        for (int nt = 0; nt < 4; ++nt) acc[mt][nt] = (v8f){};

    for (int kb = 0; kb < Kdim; kb += 32) {
        // ---- stage A tile 64x32 (one b128 per thread per matrix) ----
        {
            int m = tid >> 2, kc = (tid & 3) << 3;
            *(v8bf*)&lds_a_hi[m][kc] = *(const v8bf*)&Ahi[(size_t)(tileM + m) * lda + kb + kc];
            *(v8bf*)&lds_a_lo[m][kc] = *(const v8bf*)&Alo[(size_t)(tileM + m) * lda + kb + kc];
        }
        // ---- stage B tile 256x32 (four b128 per thread per matrix) ----
#pragma unroll
        for (int c = 0; c < 4; ++c) {
            int idx = tid + c * 256;
            int n = idx >> 2, kc = (idx & 3) << 3;
            *(v8bf*)&lds_b_hi[n][kc] = *(const v8bf*)&Bhi[(size_t)(tileN + n) * ldb + kb + kc];
            *(v8bf*)&lds_b_lo[n][kc] = *(const v8bf*)&Blo[(size_t)(tileN + n) * ldb + kb + kc];
        }
        __syncthreads();

        const int k0 = half * 8;
        v16bf a_hi[2], a_lo[2], b_hi[4], b_lo[4];
#pragma unroll
        for (int mt = 0; mt < 2; ++mt) {
            int ar = wm * 32 + mt * 16 + row;
            a_hi[mt] = make_frag(&lds_a_hi[ar][0], k0);
            a_lo[mt] = make_frag(&lds_a_lo[ar][0], k0);
        }
#pragma unroll
        for (int nt = 0; nt < 4; ++nt) {
            int br = wn * 64 + nt * 16 + row;
            b_hi[nt] = make_frag(&lds_b_hi[br][0], k0);
            b_lo[nt] = make_frag(&lds_b_lo[br][0], k0);
        }

#pragma unroll
        for (int mt = 0; mt < 2; ++mt)
#pragma unroll
            for (int nt = 0; nt < 4; ++nt) {
                acc[mt][nt] = __builtin_amdgcn_wmma_f32_16x16x32_bf16(
                    false, a_hi[mt], false, b_hi[nt], (short)0, acc[mt][nt], false, false);
                acc[mt][nt] = __builtin_amdgcn_wmma_f32_16x16x32_bf16(
                    false, a_lo[mt], false, b_hi[nt], (short)0, acc[mt][nt], false, false);
                acc[mt][nt] = __builtin_amdgcn_wmma_f32_16x16x32_bf16(
                    false, a_hi[mt], false, b_lo[nt], (short)0, acc[mt][nt], false, false);
            }
        __syncthreads();
    }

    // ---- epilogue: f32 C/D layout VGPR r -> M = r + 8*half, N = lane%16 ----
#pragma unroll
    for (int mt = 0; mt < 2; ++mt)
#pragma unroll
        for (int nt = 0; nt < 4; ++nt) {
            const int gj = tileN + wn * 64 + nt * 16 + row;
#pragma unroll
            for (int r = 0; r < 8; ++r) {
                const int gi = tileM + wm * 32 + mt * 16 + r + 8 * half;
                float v = acc[mt][nt][r];
                if (mode == 0) {
                    Cf[(size_t)gi * ldc + gj] = v;
                } else if (mode == 1) {
                    float sq = qcol[gj] - 2.0f * v + qrow[gi];
                    Cf[(size_t)gi * ldc + gj] = sqrtf(fmaxf(sq, 0.0f));
                } else {
                    size_t o = (mode == 2) ? ((size_t)gi * ldc + gj)
                                           : ((size_t)gj * ldc + gi);
                    unsigned short h = bf16_bits(v);
                    Dhi[o] = bits_to_bf16(h);
                    Dlo[o] = bits_to_bf16(bf16_bits(v - bf16_val(h)));
                }
            }
        }
}

// qm[m] = sum_d M[d,m]*H[d,m], from bf16-pair MT/HT rows (k-contiguous)
__global__ __launch_bounds__(256)
void qm_kernel(const __bf16* __restrict__ MThi, const __bf16* __restrict__ MTlo,
               const __bf16* __restrict__ HThi, const __bf16* __restrict__ HTlo,
               float* __restrict__ qm, int D) {
    int m = blockIdx.x * blockDim.x + threadIdx.x;
    size_t base = (size_t)m * D;
    float s = 0.0f;
    for (int d = 0; d < D; ++d) {
        float mv = bf_to_f(MThi[base + d]) + bf_to_f(MTlo[base + d]);
        float hv = bf_to_f(HThi[base + d]) + bf_to_f(HTlo[base + d]);
        s = fmaf(mv, hv, s);
    }
    qm[m] = s;
}

// qc[k] = sum_d C[k,d]*CG[k,d]  (fp32, rows contiguous)
__global__ __launch_bounds__(256)
void qc_kernel(const float* __restrict__ Cc, const float* __restrict__ CG,
               float* __restrict__ qc, int D) {
    int k = blockIdx.x * blockDim.x + threadIdx.x;
    float s = 0.0f;
    for (int d = 0; d < D; ++d)
        s = fmaf(Cc[(size_t)k * D + d], CG[(size_t)k * D + d], s);
    qc[k] = s;
}

extern "C" void kernel_launch(void* const* d_in, const int* in_sizes, int n_in,
                              void* d_out, int out_size, void* d_ws, size_t ws_size,
                              hipStream_t stream) {
    constexpr int N = 4096, D = 512, MC = 4096, K = 2048;

    const float* X  = (const float*)d_in[0];  // [N, D]
    const float* Mm = (const float*)d_in[1];  // [D, MC]
    const float* Ct = (const float*)d_in[2];  // [K, D]
    float* out = (float*)d_out;               // [K, MC]

    // ---- workspace carve (bf16 regions then fp32), ~33 MB ----
    __bf16* XT_hi = (__bf16*)d_ws;                    // [D][N]   2M
    __bf16* XT_lo = XT_hi + (size_t)D * N;
    __bf16* MT_hi = XT_lo + (size_t)D * N;            // [MC][D]  2M
    __bf16* MT_lo = MT_hi + (size_t)MC * D;
    __bf16* C_hi  = MT_lo + (size_t)MC * D;           // [K][D]   1M
    __bf16* C_lo  = C_hi  + (size_t)K * D;
    __bf16* G_hi  = C_lo  + (size_t)K * D;            // [D][D]   256K
    __bf16* G_lo  = G_hi  + (size_t)D * D;
    __bf16* HT_hi = G_lo  + (size_t)D * D;            // [MC][D]  2M
    __bf16* HT_lo = HT_hi + (size_t)MC * D;
    float*  CG    = (float*)(HT_lo + (size_t)MC * D); // [K][D] fp32
    float*  qm    = CG + (size_t)K * D;               // [MC]
    float*  qc    = qm + MC;                          // [K]

    dim3 blk(256);

    // 1) operand preparation: transposed/split bf16 pairs
    transpose_split_kernel<<<dim3(D / 32, N / 32), blk, 0, stream>>>(X, XT_hi, XT_lo, N, D);
    transpose_split_kernel<<<dim3(MC / 32, D / 32), blk, 0, stream>>>(Mm, MT_hi, MT_lo, D, MC);
    split_kernel<<<(K * D) / 256, blk, 0, stream>>>(Ct, C_hi, C_lo, K * D);

    // 2) G = X^T X  ->  bf16 hi/lo (mode 2).  A = XT [D][N], B^T = XT [D][N].
    gemm_bf16pair_wmma_kernel<<<dim3(D / 256, D / 64), blk, 0, stream>>>(
        XT_hi, XT_lo, N, XT_hi, XT_lo, N,
        nullptr, G_hi, G_lo, D, N, /*mode=*/2, nullptr, nullptr);

    // 3) H = G * M  ->  transposed bf16 hi/lo HT [MC][D] (mode 3).
    //    A = G [D][D], B^T = MT [MC][D].
    gemm_bf16pair_wmma_kernel<<<dim3(MC / 256, D / 64), blk, 0, stream>>>(
        G_hi, G_lo, D, MT_hi, MT_lo, D,
        nullptr, HT_hi, HT_lo, D, D, /*mode=*/3, nullptr, nullptr);

    // 4) CG = C * G  ->  fp32 (mode 0).  A = C [K][D], B^T = G^T = G (symmetric).
    gemm_bf16pair_wmma_kernel<<<dim3(D / 256, K / 64), blk, 0, stream>>>(
        C_hi, C_lo, D, G_hi, G_lo, D,
        CG, nullptr, nullptr, D, D, /*mode=*/0, nullptr, nullptr);

    // 5) qm, qc
    qm_kernel<<<MC / 256, blk, 0, stream>>>(MT_hi, MT_lo, HT_hi, HT_lo, qm, D);
    qc_kernel<<<K / 256, blk, 0, stream>>>(Ct, CG, qc, D);

    // 6) out = sqrt(max(qm[j] - 2*(C*H)[i,j] + qc[i], 0))  (mode 1, fused).
    //    A = C [K][D], B^T = HT [MC][D].
    gemm_bf16pair_wmma_kernel<<<dim3(MC / 256, K / 64), blk, 0, stream>>>(
        C_hi, C_lo, D, HT_hi, HT_lo, D,
        out, nullptr, nullptr, MC, D, /*mode=*/1, qc, qm);

    (void)in_sizes; (void)n_in; (void)out_size; (void)ws_size;
}
